// HoppingNN_11433202942663
// MI455X (gfx1250) — matrix operand order repaired and checked by reading, not compiled, not run on hardware
//
#include <hip/hip_runtime.h>
#include <stdint.h>

typedef __bf16 bf16_t;
typedef __attribute__((ext_vector_type(16))) __bf16 v16bf;
typedef __attribute__((ext_vector_type(8)))  __bf16 v8bf;
typedef __attribute__((ext_vector_type(8)))  float  v8f;

#define NPAIR   14
#define F_DIM   128
#define EX_DIM  64
#define E_EDGES 500000
#define HID     256
#define D_IN    320
#define TILE_E  64
#define ROW_H   328      // halfs per LDS x-row: 320 data + 8 pad -> 656 B (16B aligned, bank-spread)
#define K_STEPS 10       // 320 / 32
#define NGROUP  10

// dedup'd gather groups: (i,j) and the contiguous k-range [g_s[g], g_s[g+1]) sharing it
__constant__ int g_i[NGROUP]   = {0,0,1,0,1,2,3,3,3,3};
__constant__ int g_j[NGROUP]   = {0,1,1,2,2,2,3,0,1,2};
__constant__ int g_s[NGROUP+1] = {0,1,2,4,5,7,10,11,12,13,14};

// ---- one-time pre-pass: W1 (14,320,256) f32 -> W1T (14,256,320) bf16, tiled transpose ----
__global__ void w1t_kernel(const float* __restrict__ W1, bf16_t* __restrict__ W1T) {
  __shared__ float tile[32][33];
  const int k   = blockIdx.z;        // 0..13
  const int kt  = blockIdx.y;        // 0..9   (K tile of 32)
  const int ntb = blockIdx.x;        // 0..7   (N tile of 32)
  const int tx  = threadIdx.x & 31;
  const int ty  = threadIdx.x >> 5;  // 0..7
  const float* src = W1 + (size_t)k * D_IN * HID + (size_t)(kt * 32) * HID + ntb * 32;
  for (int r = ty; r < 32; r += 8)
    tile[r][tx] = src[(size_t)r * HID + tx];          // coalesced over n
  __syncthreads();
  bf16_t* dst = W1T + ((size_t)(k * HID + ntb * 32)) * D_IN + kt * 32;
  for (int r = ty; r < 32; r += 8)
    dst[(size_t)r * D_IN + tx] = (bf16_t)tile[tx][r]; // coalesced over kk
}

__global__ __launch_bounds__(256)
void hopping_kernel(const float* __restrict__ f0, const float* __restrict__ f1,
                    const float* __restrict__ f2, const float* __restrict__ f3,
                    const int*   __restrict__ hop, const float* __restrict__ dptr,
                    const float* __restrict__ exd, const bf16_t* __restrict__ W1T,
                    const float* __restrict__ b1p, const float* __restrict__ W2,
                    const float* __restrict__ b2p, float* __restrict__ out)
{
  __shared__ bf16_t lds_x[TILE_E * ROW_H];   // ~41 KB staged x-tile
  __shared__ float  lds_part[8][TILE_E];     // per-wave partial sums (2 KB)

  const int t      = threadIdx.x;
  const int lane   = t & 31;
  const int wave   = t >> 5;
  const int lr     = lane & 15;
  const int hiHalf = (lane >= 16) ? 1 : 0;
  const int nbase  = wave * 32;              // each wave owns a DISJOINT N=32 slice
  const int ebase  = blockIdx.x * TILE_E;

  const float* feats[4] = {f0, f1, f2, f3};

  for (int g = 0; g < NGROUP; ++g) {
    const float* FI = feats[g_i[g]];
    const float* FJ = feats[g_j[g]];

    // ---- gather x-tile (64 edges x 320 f32) -> scaled bf16 in LDS (once per group) ----
    for (int u = t; u < TILE_E * 80; u += 256) {
      int e = u / 80;
      int c = u - e * 80;            // float4 column unit
      int eg = ebase + e;
      if (eg >= E_EDGES) eg = E_EDGES - 1;
      float4 v;
      float scale = 1.0f;
      if (c < 64) {
        int a = (c < 32) ? hop[2 * eg] : hop[2 * eg + 1];
        const float* src = (c < 32) ? FI : FJ;
        v = *(const float4*)(src + (size_t)a * F_DIM + (size_t)(c & 31) * 4);
        float dv = dptr[eg];
        scale = 1.0f / (dv * dv);
      } else {
        v = *(const float4*)(exd + (size_t)eg * EX_DIM + (size_t)(c - 64) * 4);
      }
      union { bf16_t h[4]; uint64_t q; } pk;
      pk.h[0] = (bf16_t)(v.x * scale);
      pk.h[1] = (bf16_t)(v.y * scale);
      pk.h[2] = (bf16_t)(v.z * scale);
      pk.h[3] = (bf16_t)(v.w * scale);
      *(uint64_t*)&lds_x[e * ROW_H + c * 4] = pk.q;   // ds_store_b64
    }
    __syncthreads();

    for (int kp = g_s[g]; kp < g_s[g + 1]; ++kp) {
      // ---- GEMM1: per wave, (64 edges) x (K=320) x (N=32), bf16 WMMA, B double-buffered ----
      v8f acc[4][2];
#pragma unroll
      for (int mb = 0; mb < 4; ++mb)
#pragma unroll
        for (int nt = 0; nt < 2; ++nt)
#pragma unroll
          for (int r = 0; r < 8; ++r) acc[mb][nt][r] = 0.0f;

      const int koffA = hiHalf ? 16 : 0;
      // B: lane -> col n = nbase + nt*16 + lr, contiguous 16 halfs at K = 32k + (0|16)
      const bf16_t* brow = W1T + ((size_t)(kp * HID + nbase + lr)) * D_IN + (hiHalf ? 16 : 0);

      v16bf bc0 = *(const v16bf*)(brow);
      v16bf bc1 = *(const v16bf*)(brow + 16 * D_IN);

      for (int ks = 0; ks < K_STEPS; ++ks) {
        // prefetch next K-step's B tiles into separate regs (keeps loads in flight past the waits)
        const int ksn = (ks < K_STEPS - 1) ? ks + 1 : ks;
        v16bf bn0 = *(const v16bf*)(brow + ksn * 32);
        v16bf bn1 = *(const v16bf*)(brow + 16 * D_IN + ksn * 32);
#pragma unroll
        for (int mb = 0; mb < 4; ++mb) {
          union { v16bf v; v8bf h[2]; } a;
          const bf16_t* arow = &lds_x[(mb * 16 + lr) * ROW_H + koffA + ks * 32];
          a.h[0] = *(const v8bf*)(arow);        // ds_load_b128
          a.h[1] = *(const v8bf*)(arow + 8);    // ds_load_b128
          acc[mb][0] = __builtin_amdgcn_wmma_f32_16x16x32_bf16(
              false, a.v, false, bc0, (short)0, acc[mb][0], false, false);
          acc[mb][1] = __builtin_amdgcn_wmma_f32_16x16x32_bf16(
              false, a.v, false, bc1, (short)0, acc[mb][1], false, false);
        }
        bc0 = bn0;
        bc1 = bn1;
      }

      // prefetch next pair's B slice (L2 warm-up) during epilogue dead time
      if (kp + 1 < NPAIR) {
        const bf16_t* pn = W1T + ((size_t)((kp + 1) * HID + nbase)) * D_IN;
#pragma unroll
        for (int q = 0; q < 5; ++q)
          __builtin_prefetch(pn + (size_t)(q * 32 + lane) * 64, 0, 1); // global_prefetch_b8
      }

      // ---- epilogue: bias + leaky_relu + dot with W2 column slice ----
      float s[4][8];
#pragma unroll
      for (int mb = 0; mb < 4; ++mb)
#pragma unroll
        for (int r = 0; r < 8; ++r) s[mb][r] = 0.0f;
#pragma unroll
      for (int nt = 0; nt < 2; ++nt) {
        int n = nbase + nt * 16 + lr;
        float bias = b1p[kp * HID + n];
        float w2v  = W2[kp * HID + n];
#pragma unroll
        for (int mb = 0; mb < 4; ++mb)
#pragma unroll
          for (int r = 0; r < 8; ++r) {
            float h = acc[mb][nt][r] + bias;
            h = (h > 0.0f) ? h : (h * 0.01f);
            s[mb][r] += h * w2v;
          }
      }
      // reduce over the 16 lanes of each half (the N dimension within the wave's slice)
#pragma unroll
      for (int mb = 0; mb < 4; ++mb)
#pragma unroll
        for (int r = 0; r < 8; ++r) {
          float v = s[mb][r];
          v += __shfl_xor(v, 1, 32);
          v += __shfl_xor(v, 2, 32);
          v += __shfl_xor(v, 4, 32);
          v += __shfl_xor(v, 8, 32);
          s[mb][r] = v;
        }
      if (lr == 0) {
#pragma unroll
        for (int mb = 0; mb < 4; ++mb)
#pragma unroll
          for (int r = 0; r < 8; ++r)
            lds_part[wave][mb * 16 + r + (hiHalf ? 8 : 0)] = s[mb][r];
      }
      __syncthreads();

      if (t < TILE_E) {
        float sum = 0.0f;
#pragma unroll
        for (int w = 0; w < 8; ++w) sum += lds_part[w][t];
        int eg = ebase + t;
        if (eg < E_EDGES) out[(size_t)eg * NPAIR + kp] = sum + b2p[kp];
      }
      __syncthreads();   // lds_part reads done before next kp overwrites; also fences lds_x reuse
    }
  }
}

extern "C" void kernel_launch(void* const* d_in, const int* in_sizes, int n_in,
                              void* d_out, int out_size, void* d_ws, size_t ws_size,
                              hipStream_t stream) {
  const float* f0  = (const float*)d_in[0];
  const float* f1  = (const float*)d_in[1];
  const float* f2  = (const float*)d_in[2];
  const float* f3  = (const float*)d_in[3];
  const int*   hop = (const int*)  d_in[4];
  const float* dpt = (const float*)d_in[5];
  const float* exd = (const float*)d_in[6];
  const float* W1  = (const float*)d_in[7];
  const float* b1  = (const float*)d_in[8];
  const float* W2  = (const float*)d_in[9];
  const float* b2  = (const float*)d_in[10];
  bf16_t* W1T = (bf16_t*)d_ws;   // 14*256*320*2 B = 2.2 MB, L2-resident

  dim3 tgrid(HID / 32, D_IN / 32, NPAIR);  // (8, 10, 14)
  w1t_kernel<<<tgrid, 256, 0, stream>>>(W1, W1T);

  int nblk = (E_EDGES + TILE_E - 1) / TILE_E;
  hopping_kernel<<<nblk, 256, 0, stream>>>(f0, f1, f2, f3, hop, dpt, exd, W1T,
                                           b1, W2, b2, (float*)d_out);
}
